// BDH_80161269613186
// MI455X (gfx1250) — compile-verified, matrix operand order checked by reading, not run on hardware
//
#include <hip/hip_runtime.h>

typedef unsigned short u16;
typedef __attribute__((ext_vector_type(16))) __bf16 v16bf;
typedef __attribute__((ext_vector_type(8)))  __bf16 v8bf;
typedef __attribute__((ext_vector_type(8)))  float  v8f;
typedef __attribute__((ext_vector_type(4)))  unsigned int v4u;
typedef __attribute__((ext_vector_type(8)))  int v8i;
typedef __attribute__((ext_vector_type(4)))  int v4i;

#define LDK 40   // LDS row pitch in elements (80B: 16B-aligned, conflict-staggered)

#if defined(__AMDGCN__) && __has_builtin(__builtin_amdgcn_tensor_load_to_lds)
#define HAVE_TDM 1
#endif

__device__ __forceinline__ u16 f2bf(float f) {
  unsigned u = __builtin_bit_cast(unsigned, f);
  u += 0x7fffu + ((u >> 16) & 1u);
  return (u16)(u >> 16);
}
__device__ __forceinline__ float bf2f(u16 h) {
  unsigned u = ((unsigned)h) << 16;
  return __builtin_bit_cast(float, u);
}

#if defined(HAVE_TDM)
// One TDM descriptor moves a full 128x32 bf16 tile (row stride `ld` elements) from
// global memory into LDS at byte offset lds_off, inserting 16B of padding after every
// 64B row so the tile lands with the LDK=40-element pitch. Issued per-wave (EXEC
// ignored); tracked by TENSORcnt.
__device__ __forceinline__ void tdm_load_tile(const u16* src, unsigned lds_off,
                                              int row0, int col0, int ld) {
  unsigned long long ga =
      (unsigned long long)(size_t)(src + (size_t)row0 * ld + col0);
  v4u g0;
  g0[0] = 1u;                                    // count=1, user descriptor
  g0[1] = lds_off;                               // lds_addr (bytes)
  g0[2] = (unsigned)(ga & 0xffffffffu);          // global_addr[31:0]
  g0[3] = (unsigned)((ga >> 32) & 0x01ffffffu)   // global_addr[56:32]
        | 0x80000000u;                           // type = 2 ("image")
  v8i g1;
  g1[0] = (int)((1u << 16)      // data_size = 2 bytes
              | (1u << 20)      // pad_enable
              | (3u << 22)      // pad_interval = 16 DWORDs (one 64B row)
              | (3u << 25));    // pad_amount   = 4 DWORDs (16B)
  g1[1] = (int)(32u << 16);     // tensor_dim0 = 32 elements
  g1[2] = (int)(128u << 16);    // tensor_dim1 = 128 rows
  g1[3] = (int)(32u << 16);     // tile_dim0 = 32
  g1[4] = 128;                  // tile_dim1 = 128
  g1[5] = ld;                   // tensor_dim0_stride (elements)
  g1[6] = 0;
  g1[7] = 0;
  v4i g2 = {0, 0, 0, 0};
  v4i g3 = {0, 0, 0, 0};
#if __clang_major__ >= 23
  v8i g4 = {0, 0, 0, 0, 0, 0, 0, 0};
  __builtin_amdgcn_tensor_load_to_lds(g0, g1, g2, g3, g4, 0);
#else
  __builtin_amdgcn_tensor_load_to_lds(g0, g1, g2, g3, 0);
#endif
}
#endif

// Non-transposed tile staging: TDM (one descriptor per tile, wave 0 only) when
// available, else per-lane global_load_async_to_lds_b128 (ASYNCcnt).
__device__ __forceinline__ void stage_nt(u16* __restrict__ dst, unsigned dst_off,
                                         const u16* __restrict__ src,
                                         int row0, int col0, int ld,
                                         int tid, int wid) {
#if defined(HAVE_TDM)
  if (wid == 0) tdm_load_tile(src, dst_off, row0, col0, ld);
#else
  (void)wid; (void)dst_off;
#pragma unroll
  for (int j = 0; j < 2; ++j) {
    int chunk = tid + j * 256;          // 512 chunks of 8 elements
    int r = chunk >> 2;
    int c = (chunk & 3) * 8;
    const u16* g = src + (size_t)(row0 + r) * ld + col0 + c;
    unsigned l = (unsigned)(size_t)(dst + r * LDK + c);
    asm volatile("global_load_async_to_lds_b128 %0, %1, off"
                 :: "v"(l), "v"(g) : "memory");
  }
#endif
}

// Drain the staging engine so the *current* tile is resident, leaving the
// next tile's transfers (2 TDM ops / 4 async ops per wave) in flight.
__device__ __forceinline__ void stage_nt_fence(bool more_in_flight) {
#if defined(HAVE_TDM)
  if (more_in_flight) __builtin_amdgcn_s_wait_tensorcnt(2);
  else                __builtin_amdgcn_s_wait_tensorcnt(0);
#else
  if (more_in_flight) asm volatile("s_wait_asynccnt 0x4" ::: "memory");
  else                asm volatile("s_wait_asynccnt 0x0" ::: "memory");
#endif
}

// Transposed staging: element (r,c) = src[(col0+c)*ld + row0 + r]
// (coalesced 16B reads, scalar LDS scatter; synchronous, DScnt + barrier).
__device__ __forceinline__ void stage_tr(u16* __restrict__ dst,
                                         const u16* __restrict__ src,
                                         int row0, int col0, int ld, int tid) {
  int c  = tid >> 3;                    // 0..31 : k column
  int r0 = (tid & 7) * 16;              // 16 rows per thread
  const u16* g = src + (size_t)(col0 + c) * ld + row0 + r0;
  v8bf t0 = *(const v8bf*)(g);
  v8bf t1 = *(const v8bf*)(g + 8);
  const u16* p0 = (const u16*)&t0;
  const u16* p1 = (const u16*)&t1;
#pragma unroll
  for (int j = 0; j < 8; ++j) {
    dst[(r0 + j) * LDK + c]     = p0[j];
    dst[(r0 + 8 + j) * LDK + c] = p1[j];
  }
}

// Load one 16x32 (A) / 32x16 (B) bf16 fragment for lane `lane`, logical row `row`.
// Per ISA layout: VGPRs 0-3 hold K = 8*(lane>>4)+0..7, VGPRs 4-7 hold K = 16+8*(lane>>4)+0..7.
__device__ __forceinline__ v16bf load_frag(const u16* __restrict__ lds, int row, int lane) {
  const u16* p = lds + row * LDK + (lane >> 4) * 8;
  v8bf c0 = *(const v8bf*)(p);
  v8bf c1 = *(const v8bf*)(p + 16);
  return __builtin_shufflevector(c0, c1, 0,1,2,3,4,5,6,7,8,9,10,11,12,13,14,15);
}

// C[M,N] = epi( A x B ), bf16 operands, fp32 accumulate via v_wmma_f32_16x16x32_bf16.
// TRANS=0: A row-major [M,K], B stored [N,K]  (TDM/async staged, double-buffered)
// TRANS=1: A(m,k)=A[k*lda+m], Bt(n,k)=B[k*ldb+n]  (manual transposing stage)
// EPI: 0 = store bf16; 1 = relu->bf16; 2 = relu*X->bf16; 3 = store fp32
template <int EPI, int TRANS>
__global__ __launch_bounds__(256) void gemm_bf16(
    const u16* __restrict__ Ab, size_t strA,
    const u16* __restrict__ Bb, size_t strB,
    void* __restrict__ Cb, size_t strC,
    const u16* __restrict__ Xb, size_t strX,
    int lda, int ldb, int ldc, int K) {
  __shared__ u16 sA[2][128 * LDK];
  __shared__ u16 sB[2][128 * LDK];

  const int tid  = threadIdx.x;
  const int lane = tid & 31;
  const int wid  = tid >> 5;
  const int wm   = wid & 3;   // 4 waves along M (4*32 = 128)
  const int wn   = wid >> 2;  // 2 waves along N (2*64 = 128)
  const int m0   = blockIdx.y * 128;
  const int n0   = blockIdx.x * 128;
  const int bz   = blockIdx.z;

  const u16* A = Ab + (size_t)bz * strA;
  const u16* B = Bb + (size_t)bz * strB;

  const unsigned offA[2] = {(unsigned)(size_t)&sA[0][0], (unsigned)(size_t)&sA[1][0]};
  const unsigned offB[2] = {(unsigned)(size_t)&sB[0][0], (unsigned)(size_t)&sB[1][0]};

  v8f acc[2][4] = {};

  const int nk = K / 32;
  if (TRANS == 0) {
    stage_nt(sA[0], offA[0], A, m0, 0, lda, tid, wid);
    stage_nt(sB[0], offB[0], B, n0, 0, ldb, tid, wid);
  } else {
    stage_tr(sA[0], A, m0, 0, lda, tid);
    stage_tr(sB[0], B, n0, 0, ldb, tid);
  }

  for (int i = 0; i < nk; ++i) {
    const int cur = i & 1;
    if (TRANS == 0) {
      if (i + 1 < nk) {
        stage_nt(sA[cur ^ 1], offA[cur ^ 1], A, m0, (i + 1) * 32, lda, tid, wid);
        stage_nt(sB[cur ^ 1], offB[cur ^ 1], B, n0, (i + 1) * 32, ldb, tid, wid);
        stage_nt_fence(true);
      } else {
        stage_nt_fence(false);
      }
    } else {
      if (i + 1 < nk) {
        stage_tr(sA[cur ^ 1], A, m0, (i + 1) * 32, lda, tid);
        stage_tr(sB[cur ^ 1], B, n0, (i + 1) * 32, ldb, tid);
      }
    }
    __syncthreads();

    v16bf fa[2], fb[4];
#pragma unroll
    for (int t = 0; t < 2; ++t)
      fa[t] = load_frag(sA[cur], wm * 32 + t * 16 + (lane & 15), lane);
#pragma unroll
    for (int u = 0; u < 4; ++u)
      fb[u] = load_frag(sB[cur], wn * 64 + u * 16 + (lane & 15), lane);

#pragma unroll
    for (int t = 0; t < 2; ++t)
#pragma unroll
      for (int u = 0; u < 4; ++u)
        acc[t][u] = __builtin_amdgcn_wmma_f32_16x16x32_bf16(
            false, fa[t], false, fb[u], (short)0, acc[t][u], false, false);
    __syncthreads();   // all waves done reading buf[cur] before it is re-staged
  }

  // Epilogue: C layout — VGPR r, lane L -> m = r + 8*(L>>4), n = L&15 within 16x16 tile.
#pragma unroll
  for (int t = 0; t < 2; ++t) {
#pragma unroll
    for (int u = 0; u < 4; ++u) {
      const int mb = m0 + wm * 32 + t * 16 + (lane >> 4) * 8;
      const int n  = n0 + wn * 64 + u * 16 + (lane & 15);
#pragma unroll
      for (int r = 0; r < 8; ++r) {
        float v = acc[t][u][r];
        const size_t off = (size_t)(mb + r) * ldc + n;
        if (EPI == 0) {
          ((u16*)Cb)[(size_t)bz * strC + off] = f2bf(v);
        } else if (EPI == 1) {
          ((u16*)Cb)[(size_t)bz * strC + off] = f2bf(fmaxf(v, 0.f));
        } else if (EPI == 2) {
          float xv = bf2f(Xb[(size_t)bz * strX + off]);
          ((u16*)Cb)[(size_t)bz * strC + off] = f2bf(fmaxf(v, 0.f) * xv);
        } else {
          ((float*)Cb)[(size_t)bz * strC + off] = v;
        }
      }
    }
  }
}

__global__ __launch_bounds__(256) void embed_kernel(const int* __restrict__ idx,
                                                    const float* __restrict__ emb,
                                                    float* __restrict__ vf,
                                                    u16* __restrict__ vh) {
  const int D = 1024;
  const int row = blockIdx.x;
  const int tid = threadIdx.x;
  const int t = idx[row];
  const float* e = emb + (size_t)t * D;
#pragma unroll
  for (int j = 0; j < 4; ++j) {
    float f = e[tid + j * 256];
    vf[(size_t)row * D + tid + j * 256] = f;
    vh[(size_t)row * D + tid + j * 256] = f2bf(f);
  }
}

// src fp32 [R,C] -> dst bf16 [C,R] (tiled 32x32 via LDS, conflict-free)
__global__ __launch_bounds__(256) void transpose_cvt_kernel(const float* __restrict__ src,
                                                            u16* __restrict__ dst,
                                                            int R, int C) {
  __shared__ float t[32][33];
  const int bx = blockIdx.x * 32;  // C offset
  const int by = blockIdx.y * 32;  // R offset
  const int lx = threadIdx.x & 31;
  const int ly = threadIdx.x >> 5; // 0..7
#pragma unroll
  for (int j = 0; j < 4; ++j)
    t[ly + j * 8][lx] = src[(size_t)(by + ly + j * 8) * C + bx + lx];
  __syncthreads();
#pragma unroll
  for (int j = 0; j < 4; ++j)
    dst[(size_t)(bx + ly + j * 8) * R + by + lx] = f2bf(t[lx][ly + j * 8]);
}

// v = ln(v + ln(z)) over last dim D=1024 (no affine), eps=1e-5. Updates fp32 + bf16 copies.
__global__ __launch_bounds__(256) void ln_residual_kernel(const float* __restrict__ z,
                                                          float* __restrict__ vf,
                                                          u16* __restrict__ vh) {
  const int D = 1024;
  const int row = blockIdx.x;
  const int tid = threadIdx.x;
  const float* zr = z + (size_t)row * D;
  float* vr = vf + (size_t)row * D;
  u16* hr = vh + (size_t)row * D;
  __shared__ float rs[256], rq[256];

  float lz[4], s = 0.f, q = 0.f;
#pragma unroll
  for (int j = 0; j < 4; ++j) {
    lz[j] = zr[tid + j * 256];
    s += lz[j];
    q += lz[j] * lz[j];
  }
  rs[tid] = s; rq[tid] = q;
  __syncthreads();
  for (int o = 128; o > 0; o >>= 1) {
    if (tid < o) { rs[tid] += rs[tid + o]; rq[tid] += rq[tid + o]; }
    __syncthreads();
  }
  float mean = rs[0] * (1.f / D);
  float var  = rq[0] * (1.f / D) - mean * mean;
  float inv  = rsqrtf(var + 1e-5f);
  __syncthreads();

  float tv[4]; s = 0.f; q = 0.f;
#pragma unroll
  for (int j = 0; j < 4; ++j) {
    float t = vr[tid + j * 256] + (lz[j] - mean) * inv;
    tv[j] = t;
    s += t;
    q += t * t;
  }
  rs[tid] = s; rq[tid] = q;
  __syncthreads();
  for (int o = 128; o > 0; o >>= 1) {
    if (tid < o) { rs[tid] += rs[tid + o]; rq[tid] += rq[tid + o]; }
    __syncthreads();
  }
  mean = rs[0] * (1.f / D);
  var  = rq[0] * (1.f / D) - mean * mean;
  inv  = rsqrtf(var + 1e-5f);
#pragma unroll
  for (int j = 0; j < 4; ++j) {
    float o = (tv[j] - mean) * inv;
    vr[tid + j * 256] = o;
    hr[tid + j * 256] = f2bf(o);
  }
}

extern "C" void kernel_launch(void* const* d_in, const int* in_sizes, int n_in,
                              void* d_out, int out_size, void* d_ws, size_t ws_size,
                              hipStream_t stream) {
  const int Bc = 2, Sc = 2048, Dc = 1024, Nc = 4096, Vc = 32000, Lc = 6;
  const int*   input_ = (const int*)d_in[0];
  const float* emb    = (const float*)d_in[1];
  const float* Dx     = (const float*)d_in[2];
  const float* Dy     = (const float*)d_in[3];
  const float* E      = (const float*)d_in[4];
  const float* RO     = (const float*)d_in[5];

  char* w = (char*)d_ws;
  auto carve = [&](size_t bytes) {
    void* p = (void*)w;
    w += (bytes + 255) & ~(size_t)255;
    return p;
  };
  float* vf  = (float*)carve((size_t)Bc * Sc * Dc * 4);
  u16*   vh  = (u16*)carve((size_t)Bc * Sc * Dc * 2);
  u16*   xh  = (u16*)carve((size_t)Bc * Sc * Nc * 2);
  u16*   ah  = (u16*)carve((size_t)Bc * Sc * Dc * 2);
  u16*   yh  = (u16*)carve((size_t)Bc * Sc * Nc * 2);
  u16*   kvT = (u16*)carve((size_t)Bc * Dc * Nc * 2);  // kv^T: [B, D, N]
  float* zf  = (float*)carve((size_t)Bc * Sc * Dc * 4);
  u16*   DxT = (u16*)carve((size_t)Nc * Dc * 2);       // Dx^T: [N, D]
  u16*   DyT = (u16*)carve((size_t)Nc * Dc * 2);       // Dy^T: [N, D]
  u16*   ET  = (u16*)carve((size_t)Dc * Nc * 2);       // E^T:  [D, N]
  u16*   RT  = (u16*)carve((size_t)Vc * Dc * 2);       // readout^T: [V, D]

  transpose_cvt_kernel<<<dim3(Nc / 32, Dc / 32), 256, 0, stream>>>(Dx, DxT, Dc, Nc);
  transpose_cvt_kernel<<<dim3(Nc / 32, Dc / 32), 256, 0, stream>>>(Dy, DyT, Dc, Nc);
  transpose_cvt_kernel<<<dim3(Dc / 32, Nc / 32), 256, 0, stream>>>(E, ET, Nc, Dc);
  transpose_cvt_kernel<<<dim3(Vc / 32, Dc / 32), 256, 0, stream>>>(RO, RT, Dc, Vc);
  embed_kernel<<<Bc * Sc, 256, 0, stream>>>(input_, emb, vf, vh);

  dim3 blk(256);
  for (int l = 0; l < Lc; ++l) {
    // x = relu(v @ Dx)                      [S,N]   (TDM-staged)
    gemm_bf16<1, 0><<<dim3(Nc / 128, Sc / 128, Bc), blk, 0, stream>>>(
        vh, (size_t)Sc * Dc, DxT, 0, xh, (size_t)Sc * Nc, nullptr, 0,
        Dc, Dc, Nc, Dc);
    // kv^T = v^T @ x                        [D,N]   (both operands transposed access)
    gemm_bf16<0, 1><<<dim3(Nc / 128, Dc / 128, Bc), blk, 0, stream>>>(
        vh, (size_t)Sc * Dc, xh, (size_t)Sc * Nc, kvT, (size_t)Dc * Nc, nullptr, 0,
        Dc, Nc, Nc, Sc);
    // a = x @ kv                            [S,D]   (B = kv^T rows, TDM-staged)
    gemm_bf16<0, 0><<<dim3(Dc / 128, Sc / 128, Bc), blk, 0, stream>>>(
        xh, (size_t)Sc * Nc, kvT, (size_t)Dc * Nc, ah, (size_t)Sc * Dc, nullptr, 0,
        Nc, Nc, Dc, Nc);
    // y = relu(a @ Dy) * x                  [S,N]
    gemm_bf16<2, 0><<<dim3(Nc / 128, Sc / 128, Bc), blk, 0, stream>>>(
        ah, (size_t)Sc * Dc, DyT, 0, yh, (size_t)Sc * Nc, xh, (size_t)Sc * Nc,
        Dc, Dc, Nc, Dc);
    // z = y @ E (fp32)                      [S,D]
    gemm_bf16<3, 0><<<dim3(Dc / 128, Sc / 128, Bc), blk, 0, stream>>>(
        yh, (size_t)Sc * Nc, ET, 0, zf, (size_t)Sc * Dc, nullptr, 0,
        Nc, Nc, Dc, Nc);
    // v = ln(v + ln(z))
    ln_residual_kernel<<<Bc * Sc, 256, 0, stream>>>(zf, vf, vh);
  }
  // logits = v @ readout (fp32)             [S,V]
  gemm_bf16<3, 0><<<dim3(Vc / 128, Sc / 128, Bc), blk, 0, stream>>>(
      vh, (size_t)Sc * Dc, RT, 0, (float*)d_out, (size_t)Sc * Vc, nullptr, 0,
      Dc, Dc, Vc, Dc);
}
